// HawkesKT_21320217657960
// MI455X (gfx1250) — matrix operand order repaired and checked by reading, not compiled, not run on hardware
//
#include <hip/hip_runtime.h>
#include <hip/hip_bf16.h>

typedef __attribute__((ext_vector_type(16))) _Float16 v16h;
typedef __attribute__((ext_vector_type(8)))  _Float16 v8h;
typedef __attribute__((ext_vector_type(4)))  _Float16 v4h;
typedef __attribute__((ext_vector_type(8)))  float    v8f;
typedef __attribute__((ext_vector_type(4)))  float    v4f;

#define BATCH 32
#define SEQ   1024
#define DIM   64
#define NC    256
#define TILE  16
#define LSTR  72   // padded LDS row stride in halves (144B: 16B-aligned, destripes banks)

// Build a 16x16x32-f16 WMMA operand fragment (v16h) from an LDS row.
// Per CDNA5 ISA layout, each lane's 16 halves are two contiguous 8-half runs.
__device__ __forceinline__ v16h make_frag(const _Float16* row, int off_lo, int off_hi) {
  v8h lo = *(const v8h*)(row + off_lo);
  v8h hi = *(const v8h*)(row + off_hi);
  v16h r;
#pragma unroll
  for (int e = 0; e < 8; ++e) { r[e] = lo[e]; r[8 + e] = hi[e]; }
  return r;
}

__device__ __forceinline__ v8f wmma_f16(v16h a, v16h b, v8f c) {
  // 8 args: (neg_a, A, neg_b, B, c_mod, C, reuse_a, reuse_b)
  return __builtin_amdgcn_wmma_f32_16x16x32_f16(false, a, false, b, (short)0, c,
                                                false, false);
}

__global__ __launch_bounds__(32)
void hawkes_wmma_kernel(const int* __restrict__ concept_seq,
                        const int* __restrict__ question_seq,
                        const int* __restrict__ correctness_seq,
                        const int* __restrict__ time_seq,
                        const float* __restrict__ alpha_inter,
                        const float* __restrict__ alpha_concept,
                        const float* __restrict__ beta_inter,
                        const float* __restrict__ beta_concept,
                        const float* __restrict__ q_bias,
                        const float* __restrict__ c_bias,
                        float* __restrict__ out)
{
  __shared__ __align__(16) _Float16 sAa[TILE][LSTR];  // alpha_inter rows (sources i)
  __shared__ __align__(16) _Float16 sAb[TILE][LSTR];  // beta_inter rows
  __shared__ __align__(16) _Float16 sBa[TILE][LSTR];  // alpha_concept rows (targets j)
  __shared__ __align__(16) _Float16 sBb[TILE][LSTR];  // beta_concept rows
  __shared__ float tcol[TILE];
  __shared__ float trow[TILE];

  const int lane = threadIdx.x;         // 0..31 (wave32)
  const int b    = blockIdx.y;          // batch
  const int j0   = blockIdx.x * TILE;   // target column tile base

  const int nrow = lane & 15;           // row/col index within tile this lane owns
  const int half = lane >> 4;           // 0 or 1 (lane half)

  // ---- stage target (B) tiles once: f32 -> f16 into LDS ----
  {
    const int sel = half;               // lanes 0-15: alpha, lanes 16-31: beta
    const int c   = concept_seq[b * SEQ + j0 + nrow];
    const float* src = sel ? (beta_concept + c * DIM) : (alpha_concept + c * DIM);
    _Float16*    dst = sel ? sBb[nrow] : sBa[nrow];
#pragma unroll
    for (int k = 0; k < DIM; k += 4) {
      v4f v = *(const v4f*)(src + k);
      v4h h;
      h[0] = (_Float16)v[0]; h[1] = (_Float16)v[1];
      h[2] = (_Float16)v[2]; h[3] = (_Float16)v[3];
      *(v4h*)(dst + k) = h;
    }
    if (!sel) tcol[nrow] = (float)time_seq[b * SEQ + j0 + nrow];
  }
  __syncthreads();

  // ---- persistent B fragments (K blocks 0:31 and 32:63) ----
  // 16-bit B layout: lane (n, half) holds K = 32*kb + {8*half+[0..7], 16+8*half+[0..7]}.
  const v16h bA0 = make_frag(sBa[nrow], 8 * half,      16 + 8 * half);
  const v16h bA1 = make_frag(sBa[nrow], 32 + 8 * half, 48 + 8 * half);
  const v16h bB0 = make_frag(sBb[nrow], 8 * half,      16 + 8 * half);
  const v16h bB1 = make_frag(sBb[nrow], 32 + 8 * half, 48 + 8 * half);
  const float tj = tcol[nrow];

  const float inv_log5 = 0.6213349345596119f;  // 1/ln(5)
  const int   mb = half * 8;                   // C-matrix row base for this lane half
  float colsum = 0.0f;

  for (int i0 = 0; i0 <= j0; i0 += TILE) {
    __syncthreads();  // previous iteration's LDS reads complete before overwrite
    // ---- stage source (A) tiles: interaction embeddings ----
    {
      const int sel   = half;
      const int s     = b * SEQ + i0 + nrow;
      const int inter = concept_seq[s] + NC * correctness_seq[s];
      const float* src = sel ? (beta_inter + inter * DIM) : (alpha_inter + inter * DIM);
      _Float16*    dst = sel ? sAb[nrow] : sAa[nrow];
#pragma unroll
      for (int k = 0; k < DIM; k += 4) {
        v4f v = *(const v4f*)(src + k);
        v4h h;
        h[0] = (_Float16)v[0]; h[1] = (_Float16)v[1];
        h[2] = (_Float16)v[2]; h[3] = (_Float16)v[3];
        *(v4h*)(dst + k) = h;
      }
      if (!sel) trow[nrow] = (float)time_seq[s];
      if (i0 + TILE <= j0)  // hint next tile's index loads toward the caches
        __builtin_prefetch(&concept_seq[s + TILE], 0, 3);
    }
    __syncthreads();

    // ---- A fragments ----
    // 16-bit A layout: lane (m, half) holds K = 32*kb + {8*half+[0..7], 16+8*half+[0..7]}.
    const v16h aA0 = make_frag(sAa[nrow], 8 * half,      16 + 8 * half);
    const v16h aA1 = make_frag(sAa[nrow], 32 + 8 * half, 48 + 8 * half);
    const v16h aB0 = make_frag(sAb[nrow], 8 * half,      16 + 8 * half);
    const v16h aB1 = make_frag(sAb[nrow], 32 + 8 * half, 48 + 8 * half);

    // ---- matrix cores: alphas and betas 16x16 tiles, K=64 via 2 x K=32 ----
    v8f accA = {};
    accA = wmma_f16(aA0, bA0, accA);
    accA = wmma_f16(aA1, bA1, accA);
    v8f accB = {};
    accB = wmma_f16(aB0, bB0, accB);
    accB = wmma_f16(aB1, bB1, accB);

    // ---- fused elementwise + causal reduction over sources ----
    // C layout: VGPR r -> global row i = i0 + mb + r; column j = j0 + nrow.
    const int gj = j0 + nrow;
#pragma unroll
    for (int r = 0; r < 8; ++r) {
      const int   gi = i0 + mb + r;
      const float ti = trow[mb + r];
      const float dl = __logf(fabsf(ti - tj) + 1e-10f) * inv_log5;
      float beta = accB[r] + 1.0f;
      beta = fminf(fmaxf(beta, 0.0f), 10.0f);
      const float ce = accA[r] * __expf(-beta * dl);
      colsum += (gi < gj) ? ce : 0.0f;
    }
  }

  // combine the two lane-halves that hold the same column (rows 0-7 vs 8-15)
  colsum += __shfl_xor(colsum, 16);

  if (lane < 16) {
    const int gj = j0 + nrow;
    if (gj >= 1) {
      const int q = question_seq[b * SEQ + gj];
      const int c = concept_seq[b * SEQ + gj];
      const float x = q_bias[q] + c_bias[c] + colsum;
      out[b * (SEQ - 1) + (gj - 1)] = 1.0f / (1.0f + __expf(-x));
    }
  }
}

extern "C" void kernel_launch(void* const* d_in, const int* in_sizes, int n_in,
                              void* d_out, int out_size, void* d_ws, size_t ws_size,
                              hipStream_t stream) {
  (void)in_sizes; (void)n_in; (void)d_ws; (void)ws_size; (void)out_size;
  const int*   concept_seq     = (const int*)d_in[0];
  const int*   question_seq    = (const int*)d_in[1];
  const int*   correctness_seq = (const int*)d_in[2];
  const int*   time_seq        = (const int*)d_in[3];
  const float* alpha_inter     = (const float*)d_in[4];
  const float* alpha_concept   = (const float*)d_in[5];
  const float* beta_inter      = (const float*)d_in[6];
  const float* beta_concept    = (const float*)d_in[7];
  const float* q_bias          = (const float*)d_in[8];
  const float* c_bias          = (const float*)d_in[9];
  float*       out             = (float*)d_out;

  dim3 grid(SEQ / TILE, BATCH);  // (64, 32): one wave per (target tile, batch)
  hawkes_wmma_kernel<<<grid, 32, 0, stream>>>(
      concept_seq, question_seq, correctness_seq, time_seq,
      alpha_inter, alpha_concept, beta_inter, beta_concept,
      q_bias, c_bias, out);
}
